// Pointnet2_44109314130658
// MI455X (gfx1250) — compile-verified, matrix operand order checked by reading, not run on hardware
//
#include <hip/hip_runtime.h>

// ---------------------------------------------------------------------------
// Types for CDNA5 WMMA (wave32): v_wmma_f32_16x16x32_bf16
// ---------------------------------------------------------------------------
typedef __bf16 bf16;
typedef __attribute__((ext_vector_type(16))) __bf16 bf16x16;
typedef __attribute__((ext_vector_type(8)))  __bf16 bf16x8;
typedef __attribute__((ext_vector_type(8)))  float  fx8;
typedef __attribute__((ext_vector_type(4)))  unsigned int u32x4;
typedef __attribute__((ext_vector_type(8)))  int i32x8;
typedef __attribute__((ext_vector_type(4)))  int i32x4;

#define NWAVES 8   // 256-thread blocks = 8 wave32

// ---------------------------------------------------------------------------
// Fragment helpers.
// A (16x32 bf16) per ISA 7.12.2: lanes 0-15 -> M=lane, K runs {0..7},{16..23};
// lanes 16-31 -> K runs {8..15},{24..31}.  Two 16B contiguous loads per lane.
// B is stored pre-packed (pack_w_kernel) so each lane loads 16 contiguous
// bf16 (32B) per 32x16 tile.
// ---------------------------------------------------------------------------
__device__ __forceinline__ bf16x16 load_a_frag(const bf16* X, int ldw, int mt,
                                               int kt, int lane) {
  int m  = lane & 15;
  int kb = kt * 32 + ((lane >> 4) << 3);
  const bf16* p = X + (size_t)(mt * 16 + m) * ldw + kb;
  bf16x8 lo = *(const bf16x8*)(p);
  bf16x8 hi = *(const bf16x8*)(p + 16);
  return __builtin_shufflevector(lo, hi, 0,1,2,3,4,5,6,7,8,9,10,11,12,13,14,15);
}

__device__ __forceinline__ bf16x16 load_b_frag(const bf16* Wp, int ntiles,
                                               int kt, int nt, int lane) {
  const bf16* p = Wp + (((size_t)kt * ntiles + nt) * 32 + lane) * 16;
  return *(const bf16x16*)(p);
}

// One MLP layer: Y[KSxCOUT] = relu(X[KSxCIN] @ W + b), X/Y in LDS, W packed in
// global (hot in L2 across blocks).  CIN multiple of 32, COUT multiple of 16.
template<int KS, int CIN, int COUT>
__device__ __forceinline__ void mlp_layer_wmma(const bf16* X, bf16* Y,
                                               const bf16* Wp, const float* bias,
                                               int tid) {
  const int wave = tid >> 5, lane = tid & 31;
  constexpr int MT = KS / 16, NT = COUT / 16, KT = CIN / 32;
  for (int t = wave; t < MT * NT; t += NWAVES) {
    int mt = t / NT, nt = t % NT;
    fx8 acc = {};
    #pragma unroll
    for (int kt = 0; kt < KT; ++kt) {
      bf16x16 a = load_a_frag(X, CIN, mt, kt, lane);
      bf16x16 b = load_b_frag(Wp, NT, kt, nt, lane);
      acc = __builtin_amdgcn_wmma_f32_16x16x32_bf16(false, a, false, b,
                                                    (short)0, acc, false, false);
    }
    // C/D layout: lanes 0-15: N=lane, M=r; lanes 16-31: N=lane-16, M=r+8
    int n  = lane & 15;
    int mh = (lane >> 4) << 3;
    float bv = bias[nt * 16 + n];
    #pragma unroll
    for (int r = 0; r < 8; ++r) {
      float v = acc[r] + bv;
      v = v > 0.f ? v : 0.f;
      Y[(size_t)(mt * 16 + mh + r) * COUT + nt * 16 + n] = (bf16)v;
    }
  }
}

// ---------------------------------------------------------------------------
// Tensor Data Mover: 2D tile (rows x width bf16, row stride = width) from
// global memory to LDS.  D# built per CDNA5 ISA ch.8 (group0/group1), issued
// once per wave (EXEC ignored by TDM), tracked by TENSORcnt.
// 6-arg builtin variant (clang-23 / therock-10.0 headers).
// ---------------------------------------------------------------------------
__device__ __forceinline__ void tdm_load_2d_to_lds(const void* gaddr,
                                                   unsigned lds_off,
                                                   unsigned width_elems,
                                                   unsigned rows) {
  unsigned long long ga = (unsigned long long)(uintptr_t)gaddr;
  u32x4 g0;
  g0[0] = 1u;                                           // count=1 (valid), user
  g0[1] = lds_off;                                      // lds_addr (bytes)
  g0[2] = (unsigned)(ga & 0xffffffffu);                 // global_addr[31:0]
  g0[3] = (unsigned)((ga >> 32) & 0x01ffffffu) | (2u << 30); // addr[56:32]|type=2
  i32x8 g1;
  unsigned td0 = width_elems, td1 = rows;
  unsigned tl0 = width_elems, tl1 = rows;
  g1[0] = (int)(1u << 16);                              // data_size=1 (2 bytes)
  g1[1] = (int)((td0 & 0xffffu) << 16);                 // tensor_dim0[15:0]
  g1[2] = (int)((td0 >> 16) | ((td1 & 0xffffu) << 16)); // dim0 hi | dim1 lo
  g1[3] = (int)((td1 >> 16) | (tl0 << 16));             // dim1 hi | tile_dim0
  g1[4] = (int)(tl1 & 0xffffu);                         // tile_dim1, tile_dim2=0
  g1[5] = (int)width_elems;                             // tensor_dim0_stride lo
  g1[6] = 0;                                            // stride hi | dim1_stride
  g1[7] = 0;
  i32x4 z4 = {0, 0, 0, 0};
  i32x8 z8 = {0, 0, 0, 0, 0, 0, 0, 0};
  __builtin_amdgcn_tensor_load_to_lds(g0, g1, z4, z4, z8, 0);
}

// ---------------------------------------------------------------------------
// Weight packing: f32 (cin x cout) -> bf16 B-fragment layout, K padded to Kp.
// Packed element i: tile = i/512, lane = (i/16)%32, j = i%16.
// ---------------------------------------------------------------------------
__global__ __launch_bounds__(256) void pack_w_kernel(const float* __restrict__ W,
                                                     bf16* __restrict__ Wp,
                                                     int cin, int cout, int Kp) {
  int i = blockIdx.x * 256 + threadIdx.x;
  if (i >= Kp * cout) return;
  int j    = i & 15;
  int lane = (i >> 4) & 31;
  int tile = i >> 9;
  int ntiles = cout / 16;
  int kt = tile / ntiles, nt = tile % ntiles;
  int n  = nt * 16 + (lane & 15);
  int kb = (lane >> 4) << 3;
  int kl = (j < 8) ? (kb + j) : (kb + 16 + (j - 8));
  int k  = kt * 32 + kl;
  float v = (k < cin) ? W[(size_t)k * cout + n] : 0.f;
  Wp[i] = (bf16)v;
}

// ---------------------------------------------------------------------------
// Farthest point sampling: one block per batch, dist[] in LDS,
// block argmax via packed (float-bits<<32 | (N-1-j)) u64 max reduction.
// ---------------------------------------------------------------------------
__global__ __launch_bounds__(256) void fps_kernel(const float* __restrict__ xyz,
                                                  int N, int npoint,
                                                  int* __restrict__ out_idx,
                                                  float* __restrict__ new_xyz) {
  __shared__ float dist[2048];
  __shared__ unsigned long long red[256];
  int b = blockIdx.x;
  int tid = threadIdx.x;
  const float* p = xyz + (size_t)b * N * 3;
  for (int j = tid; j < N; j += 256) dist[j] = 1e10f;
  __syncthreads();
  int far = 0;
  for (int i = 0; i < npoint; ++i) {
    float cx = p[far * 3 + 0], cy = p[far * 3 + 1], cz = p[far * 3 + 2];
    if (tid == 0) {
      out_idx[(size_t)b * npoint + i] = far;
      new_xyz[((size_t)b * npoint + i) * 3 + 0] = cx;
      new_xyz[((size_t)b * npoint + i) * 3 + 1] = cy;
      new_xyz[((size_t)b * npoint + i) * 3 + 2] = cz;
    }
    unsigned long long best = 0ull;
    for (int j = tid; j < N; j += 256) {
      float dx = p[j * 3 + 0] - cx, dy = p[j * 3 + 1] - cy, dz = p[j * 3 + 2] - cz;
      float d = dx * dx + dy * dy + dz * dz;
      float m = fminf(dist[j], d);
      dist[j] = m;
      unsigned long long key =
          ((unsigned long long)__float_as_uint(m) << 32) | (unsigned)(N - 1 - j);
      best = best > key ? best : key;
    }
    __syncthreads();          // everyone done reading red[0] from prev iter
    red[tid] = best;
    __syncthreads();
    for (int s = 128; s > 0; s >>= 1) {
      if (tid < s) { unsigned long long o = red[tid + s]; if (o > red[tid]) red[tid] = o; }
      __syncthreads();
    }
    far = N - 1 - (int)(red[0] & 0xffffffffu);
  }
}

// ---------------------------------------------------------------------------
// Ball query: one wave per centroid; ballot/popc compaction picks the first K
// in-radius indices in index order (matches reference's sorted-where trick).
// ---------------------------------------------------------------------------
__global__ __launch_bounds__(256) void ball_query_kernel(
    const float* __restrict__ xyz, const float* __restrict__ new_xyz,
    int* __restrict__ out_idx, int Nsrc, int npoint, int total, int K, float r2) {
  int w = (blockIdx.x * 256 + threadIdx.x) >> 5;
  int lane = threadIdx.x & 31;
  if (w >= total) return;
  int b = w / npoint;
  const float* src = xyz + (size_t)b * Nsrc * 3;
  float cx = new_xyz[(size_t)w * 3 + 0];
  float cy = new_xyz[(size_t)w * 3 + 1];
  float cz = new_xyz[(size_t)w * 3 + 2];
  int* out = out_idx + (size_t)w * K;
  int count = 0;
  for (int base = 0; base < Nsrc && count < K; base += 32) {
    int j = base + lane;
    bool in = false;
    if (j < Nsrc) {
      float dx = src[j * 3 + 0] - cx, dy = src[j * 3 + 1] - cy, dz = src[j * 3 + 2] - cz;
      in = (dx * dx + dy * dy + dz * dz <= r2);
    }
    unsigned mask = __builtin_amdgcn_ballot_w32(in);
    int prefix = __popc(mask & ((1u << lane) - 1u));
    if (in) { int slot = count + prefix; if (slot < K) out[slot] = j; }
    count += __popc(mask);
  }
  if (count < K) {
    int fill = (count > 0) ? out[0] : (Nsrc - 1);
    for (int s = count + lane; s < K; s += 32) out[s] = fill;
  }
}

// ---------------------------------------------------------------------------
// Fused SA branch: per-centroid gather -> 3-layer bf16 WMMA MLP in LDS ->
// maxpool over K samples -> write slice of the concat feature buffer (f32).
// One block (8 wave32) per centroid.  Dynamic LDS ping-pong buffers.
// ---------------------------------------------------------------------------
template<int KS, int CF, int CINP, int C1, int C2, int C3>
__global__ __launch_bounds__(256) void sa_branch_kernel(
    const float* __restrict__ xyz, const float* __restrict__ feats,
    const float* __restrict__ new_xyz, const int* __restrict__ ball_idx,
    const bf16* __restrict__ W1, const bf16* __restrict__ W2, const bf16* __restrict__ W3,
    const float* __restrict__ b1, const float* __restrict__ b2, const float* __restrict__ b3,
    float* __restrict__ out_f, int npoint, int Nsrc, int CTOT, int ch_off) {
  constexpr int B0W = (CINP > C2) ? CINP : C2;
  constexpr int B1W = (C1 > C3) ? C1 : C3;
  extern __shared__ __align__(16) char smem_raw[];
  bf16* buf0 = (bf16*)smem_raw;
  bf16* buf1 = buf0 + (size_t)KS * B0W;
  __shared__ int sidx[KS];

  int g = blockIdx.x;                 // b*npoint + m
  int b = g / npoint;
  int tid = threadIdx.x;

  // warm L2/WGP$ for later-layer weight streams (global_prefetch_b8)
  for (size_t off = (size_t)tid * 64; off < (size_t)C1 * C2; off += 256 * 64)
    __builtin_prefetch(W2 + off, 0, 1);
  for (size_t off = (size_t)tid * 64; off < (size_t)C2 * C3; off += 256 * 64)
    __builtin_prefetch(W3 + off, 0, 1);

  if (tid < KS) sidx[tid] = ball_idx[(size_t)g * KS + tid];
  __syncthreads();

  float cx = new_xyz[(size_t)g * 3 + 0];
  float cy = new_xyz[(size_t)g * 3 + 1];
  float cz = new_xyz[(size_t)g * 3 + 2];

  // gather + bf16 convert + zero-pad into buf0 (KS x CINP)
  for (int e = tid; e < KS * CINP; e += 256) {
    int k = e / CINP, c = e % CINP;
    int si = sidx[k];
    float v = 0.f;
    if (c < CF) {
      v = feats[((size_t)b * Nsrc + si) * CF + c];
    } else if (c == CF) {
      v = xyz[((size_t)b * Nsrc + si) * 3 + 0] - cx;
    } else if (c == CF + 1) {
      v = xyz[((size_t)b * Nsrc + si) * 3 + 1] - cy;
    } else if (c == CF + 2) {
      v = xyz[((size_t)b * Nsrc + si) * 3 + 2] - cz;
    }
    buf0[e] = (bf16)v;
  }
  __syncthreads();

  mlp_layer_wmma<KS, CINP, C1>(buf0, buf1, W1, b1, tid);
  __syncthreads();
  mlp_layer_wmma<KS, C1, C2>(buf1, buf0, W2, b2, tid);
  __syncthreads();
  mlp_layer_wmma<KS, C2, C3>(buf0, buf1, W3, b3, tid);
  __syncthreads();

  // max over KS samples -> f32 concat buffer
  for (int c = tid; c < C3; c += 256) {
    float mx = -1e30f;
    for (int k = 0; k < KS; ++k) {
      float v = (float)buf1[(size_t)k * C3 + c];
      mx = mx > v ? mx : v;
    }
    out_f[(size_t)g * CTOT + ch_off + c] = mx;
  }
}

// ---------------------------------------------------------------------------
// Generic GEMM: out = relu(A[MxKp]bf16 @ Wp + b).  16-row A stripe moved into
// dynamic LDS by the Tensor Data Mover; each wave owns column tiles.
// M multiple of 16.
// ---------------------------------------------------------------------------
template<bool OUTF32>
__global__ __launch_bounds__(256) void gemm_relu_kernel(
    const bf16* __restrict__ A, const bf16* __restrict__ Wp,
    const float* __restrict__ bias, void* __restrict__ out,
    int Kp, int N, int Mvalid) {
  extern __shared__ __align__(16) char smem_raw[];
  bf16* atile = (bf16*)smem_raw;                    // 16 x Kp
  int mt0 = blockIdx.x;
  int tid = threadIdx.x;
  const bf16* Arow = A + (size_t)mt0 * 16 * Kp;

  if (tid < 32) {                                   // wave 0 issues one TDM op
    tdm_load_2d_to_lds(Arow, (unsigned)(uintptr_t)atile, (unsigned)Kp, 16u);
    __builtin_amdgcn_s_wait_tensorcnt(0);
  }
  __syncthreads();

  int wave = tid >> 5, lane = tid & 31;
  int NT = N / 16, KT = Kp / 32;
  for (int nt = wave; nt < NT; nt += NWAVES) {
    fx8 acc = {};
    for (int kt = 0; kt < KT; ++kt) {
      bf16x16 a = load_a_frag(atile, Kp, 0, kt, lane);
      bf16x16 bfrag = load_b_frag(Wp, NT, kt, nt, lane);
      acc = __builtin_amdgcn_wmma_f32_16x16x32_bf16(false, a, false, bfrag,
                                                    (short)0, acc, false, false);
    }
    int n  = lane & 15;
    int mh = (lane >> 4) << 3;
    float bv = bias[nt * 16 + n];
    #pragma unroll
    for (int r = 0; r < 8; ++r) {
      float v = acc[r] + bv;
      v = v > 0.f ? v : 0.f;
      int mrow = mt0 * 16 + mh + r;
      if (OUTF32) {
        if (mrow < Mvalid) ((float*)out)[(size_t)mrow * N + nt * 16 + n] = v;
      } else {
        ((bf16*)out)[(size_t)mrow * N + nt * 16 + n] = (bf16)v;
      }
    }
  }
}

// SA3 input build: rows = B*128, cols padded 672: [l2_xyz(3) | l2_f(640) | 0pad]
__global__ __launch_bounds__(256) void build_x3_kernel(
    const float* __restrict__ nx2, const float* __restrict__ l2f,
    bf16* __restrict__ X) {
  int i = blockIdx.x * 256 + threadIdx.x;
  if (i >= 1024 * 672) return;
  int row = i / 672, c = i % 672;
  float v = 0.f;
  if (c < 3)        v = nx2[(size_t)row * 3 + c];
  else if (c < 643) v = l2f[(size_t)row * 640 + (c - 3)];
  X[i] = (bf16)v;
}

// max over 128 points per batch; build padded 16x1024 bf16 input for fc1
__global__ __launch_bounds__(256) void maxpool3_kernel(const bf16* __restrict__ Y3,
                                                       bf16* __restrict__ Xfc) {
  int i = blockIdx.x * 256 + threadIdx.x;
  if (i >= 16 * 1024) return;
  int b = i / 1024, c = i % 1024;
  float mx = 0.f;
  if (b < 8) {
    mx = -1e30f;
    for (int m = 0; m < 128; ++m) {
      float v = (float)Y3[((size_t)(b * 128 + m)) * 1024 + c];
      mx = mx > v ? mx : v;
    }
  }
  Xfc[i] = (bf16)mx;
}

// ---------------------------------------------------------------------------
// Host orchestration
// ---------------------------------------------------------------------------
static inline void* wsalloc(char*& p, size_t bytes) {
  void* r = (void*)p;
  p += (bytes + 255) & ~(size_t)255;
  return r;
}

extern "C" void kernel_launch(void* const* d_in, const int* in_sizes, int n_in,
                              void* d_out, int out_size, void* d_ws, size_t ws_size,
                              hipStream_t stream) {
  (void)in_sizes; (void)n_in; (void)out_size; (void)ws_size;
  const int B = 8, N = 2048, NP1 = 512, NP2 = 128;
  const float* points  = (const float*)d_in[0];
  const float* normals = (const float*)d_in[1];
  auto F = [&](int i) { return (const float*)d_in[i]; };

  char* wp = (char*)d_ws;
  int*   fps1 = (int*)  wsalloc(wp, (size_t)B * NP1 * 4);
  float* nx1  = (float*)wsalloc(wp, (size_t)B * NP1 * 3 * 4);
  const int   K1[3] = {16, 32, 128};
  const float R1[3] = {0.1f, 0.2f, 0.4f};
  int* ball1[3];
  for (int i = 0; i < 3; ++i) ball1[i] = (int*)wsalloc(wp, (size_t)B * NP1 * K1[i] * 4);
  float* l1f  = (float*)wsalloc(wp, (size_t)B * NP1 * 320 * 4);
  int*   fps2 = (int*)  wsalloc(wp, (size_t)B * NP2 * 4);
  float* nx2  = (float*)wsalloc(wp, (size_t)B * NP2 * 3 * 4);
  const int   K2[3] = {32, 64, 128};
  const float R2[3] = {0.2f, 0.4f, 0.8f};
  int* ball2[3];
  for (int i = 0; i < 3; ++i) ball2[i] = (int*)wsalloc(wp, (size_t)B * NP2 * K2[i] * 4);
  float* l2f  = (float*)wsalloc(wp, (size_t)B * NP2 * 640 * 4);

  // d_in layout (setup_inputs dict insertion order): per-branch W0,W1,W2,b0,b1,b2
  struct PW { int di, cin, cout, kp; };
  const PW pws[22] = {
    {2, 6, 32, 32},    {3, 32, 32, 32},   {4, 32, 64, 32},      // sa1 b0
    {8, 6, 64, 32},    {9, 64, 64, 64},   {10, 64, 128, 64},    // sa1 b1
    {14, 6, 64, 32},   {15, 64, 96, 64},  {16, 96, 128, 96},    // sa1 b2
    {20, 323, 64, 352},{21, 64, 64, 64},  {22, 64, 128, 64},    // sa2 b0
    {26, 323, 128, 352},{27, 128, 128, 128},{28, 128, 256, 128},// sa2 b1
    {32, 323, 128, 352},{33, 128, 128, 128},{34, 128, 256, 128},// sa2 b2
    {38, 643, 256, 672},{39, 256, 512, 256},{40, 512, 1024, 512},// sa3
    {44, 1024, 512, 1024}                                        // fc1
  };
  bf16* pk[22];
  for (int i = 0; i < 22; ++i) {
    size_t elems = (size_t)pws[i].kp * pws[i].cout;
    pk[i] = (bf16*)wsalloc(wp, elems * 2);
    int grid = (int)((elems + 255) / 256);
    pack_w_kernel<<<grid, 256, 0, stream>>>(F(pws[i].di), pk[i],
                                            pws[i].cin, pws[i].cout, pws[i].kp);
  }
  bf16* X3  = (bf16*)wsalloc(wp, (size_t)1024 * 672 * 2);
  bf16* Y1  = (bf16*)wsalloc(wp, (size_t)1024 * 256 * 2);
  bf16* Y2  = (bf16*)wsalloc(wp, (size_t)1024 * 512 * 2);
  bf16* Y3  = (bf16*)wsalloc(wp, (size_t)1024 * 1024 * 2);
  bf16* Xfc = (bf16*)wsalloc(wp, (size_t)16 * 1024 * 2);

  // ---- SA1 ----
  fps_kernel<<<B, 256, 0, stream>>>(points, N, NP1, fps1, nx1);
  for (int i = 0; i < 3; ++i)
    ball_query_kernel<<<(B * NP1) / 8, 256, 0, stream>>>(
        points, nx1, ball1[i], N, NP1, B * NP1, K1[i], R1[i] * R1[i]);

  sa_branch_kernel<16, 3, 32, 32, 32, 64><<<B * NP1, 256, (16*32 + 16*64) * 2, stream>>>(
      points, normals, nx1, ball1[0], pk[0], pk[1], pk[2], F(5), F(6), F(7),
      l1f, NP1, N, 320, 0);
  sa_branch_kernel<32, 3, 32, 64, 64, 128><<<B * NP1, 256, (32*64 + 32*128) * 2, stream>>>(
      points, normals, nx1, ball1[1], pk[3], pk[4], pk[5], F(11), F(12), F(13),
      l1f, NP1, N, 320, 64);
  sa_branch_kernel<128, 3, 32, 64, 96, 128><<<B * NP1, 256, (128*96 + 128*128) * 2, stream>>>(
      points, normals, nx1, ball1[2], pk[6], pk[7], pk[8], F(17), F(18), F(19),
      l1f, NP1, N, 320, 192);

  // ---- SA2 ----
  fps_kernel<<<B, 256, 0, stream>>>(nx1, NP1, NP2, fps2, nx2);
  for (int i = 0; i < 3; ++i)
    ball_query_kernel<<<(B * NP2) / 8, 256, 0, stream>>>(
        nx1, nx2, ball2[i], NP1, NP2, B * NP2, K2[i], R2[i] * R2[i]);

  sa_branch_kernel<32, 320, 352, 64, 64, 128><<<B * NP2, 256, (32*352 + 32*128) * 2, stream>>>(
      nx1, l1f, nx2, ball2[0], pk[9], pk[10], pk[11], F(23), F(24), F(25),
      l2f, NP2, NP1, 640, 0);
  sa_branch_kernel<64, 320, 352, 128, 128, 256><<<B * NP2, 256, (64*352 + 64*256) * 2, stream>>>(
      nx1, l1f, nx2, ball2[1], pk[12], pk[13], pk[14], F(29), F(30), F(31),
      l2f, NP2, NP1, 640, 128);
  sa_branch_kernel<128, 320, 352, 128, 128, 256><<<B * NP2, 256, (128*352 + 128*256) * 2, stream>>>(
      nx1, l1f, nx2, ball2[2], pk[15], pk[16], pk[17], F(35), F(36), F(37),
      l2f, NP2, NP1, 640, 384);

  // ---- SA3 (global) ----
  build_x3_kernel<<<(1024 * 672 + 255) / 256, 256, 0, stream>>>(nx2, l2f, X3);
  gemm_relu_kernel<false><<<1024 / 16, 256, 16 * 672 * 2, stream>>>(
      X3, pk[18], F(41), (void*)Y1, 672, 256, 1024);
  gemm_relu_kernel<false><<<1024 / 16, 256, 16 * 256 * 2, stream>>>(
      Y1, pk[19], F(42), (void*)Y2, 256, 512, 1024);
  gemm_relu_kernel<false><<<1024 / 16, 256, 16 * 512 * 2, stream>>>(
      Y2, pk[20], F(43), (void*)Y3, 512, 1024, 1024);
  maxpool3_kernel<<<(16 * 1024) / 256, 256, 0, stream>>>(Y3, Xfc);

  // ---- fc1 -> d_out (f32, 8x512) ----
  gemm_relu_kernel<true><<<1, 256, 16 * 1024 * 2, stream>>>(
      Xfc, pk[21], F(45), d_out, 1024, 512, 8);
}
